// MoE_83786222011162
// MI455X (gfx1250) — compile-verified, matrix operand order checked by reading, not run on hardware
//
#include <hip/hip_runtime.h>
#include <hip/hip_bf16.h>

// MoE + residual blend for MI455X (gfx1250), wave32, bf16 WMMA path.
// d_out layout (floats): [8192*1024 output] [l_aux] [exp_counts as float x8]

typedef __bf16 bf16_t;
typedef bf16_t bf16x16 __attribute__((ext_vector_type(16)));
typedef float  f32x8   __attribute__((ext_vector_type(8)));

#define H_DIM  1024
#define FF_DIM 4096
#define NEXP   8
#define NTOK   8192
#define CAP    1024

__device__ __forceinline__ unsigned short f2bf(float f) {
  unsigned u = __float_as_uint(f);
  u += 0x7FFFu + ((u >> 16) & 1u);      // round-to-nearest-even
  return (unsigned short)(u >> 16);
}

__device__ __forceinline__ float gelu_tanh(float x) {
  float x3 = x * x * x;
  return 0.5f * x * (1.0f + tanhf(0.7978845608028654f * (x + 0.044715f * x3)));
}

// ---------------- gating: logits, softmax, argmax, coef softmax -------------
__global__ __launch_bounds__(256)
void gate_kernel(const float* __restrict__ x, const float* __restrict__ wg,
                 const float* __restrict__ coef_w, const float* __restrict__ coef_b,
                 int* __restrict__ expid, float* __restrict__ gateprob,
                 float* __restrict__ coef0, float* __restrict__ coef1,
                 float* __restrict__ me_sum) {
  const int tid = threadIdx.x, lane = tid & 31, wave = tid >> 5;
  const long t = (long)blockIdx.x * 8 + wave;
  const float* xr = x + t * H_DIM;
  float acc[10];
#pragma unroll
  for (int c = 0; c < 10; ++c) acc[c] = 0.f;
  for (int i = lane; i < H_DIM; i += 32) {
    float xv = xr[i];
    const float4 w0 = *(const float4*)(wg + (long)i * NEXP);
    const float4 w1 = *(const float4*)(wg + (long)i * NEXP + 4);
    acc[0] += xv * w0.x; acc[1] += xv * w0.y; acc[2] += xv * w0.z; acc[3] += xv * w0.w;
    acc[4] += xv * w1.x; acc[5] += xv * w1.y; acc[6] += xv * w1.z; acc[7] += xv * w1.w;
    acc[8] += xv * coef_w[i * 2 + 0];
    acc[9] += xv * coef_w[i * 2 + 1];
  }
#pragma unroll
  for (int off = 16; off >= 1; off >>= 1)
#pragma unroll
    for (int c = 0; c < 10; ++c)
      acc[c] += __shfl_xor(acc[c], off, 32);
  if (lane == 0) {
    float mx = acc[0]; int am = 0;
#pragma unroll
    for (int c = 1; c < NEXP; ++c) if (acc[c] > mx) { mx = acc[c]; am = c; }
    float p[NEXP], den = 0.f;
#pragma unroll
    for (int c = 0; c < NEXP; ++c) { p[c] = __expf(acc[c] - mx); den += p[c]; }
    float inv = 1.0f / den;
#pragma unroll
    for (int c = 0; c < NEXP; ++c) { p[c] *= inv; atomicAdd(&me_sum[c], p[c]); }
    expid[t]    = am;
    gateprob[t] = p[am];
    float c0l = acc[8] + coef_b[0], c1l = acc[9] + coef_b[1];
    float m2 = fmaxf(c0l, c1l);
    float e0 = __expf(c0l - m2), e1 = __expf(c1l - m2);
    float s = e0 + e1;
    coef0[t] = e0 / s;
    coef1[t] = e1 / s;
  }
}

// --------- deterministic capacity assignment (cumsum semantics) -------------
__global__ __launch_bounds__(256)
void assign_kernel(const int* __restrict__ expid, const float* __restrict__ me_sum,
                   int* __restrict__ tok2slot, int* __restrict__ slot2tok,
                   float* __restrict__ out_tail, int write_tail) {
  __shared__ float partial[NEXP];
  __shared__ int counts[NEXP];
  const int tid = threadIdx.x, lane = tid & 31, e = tid >> 5;   // wave e owns expert e
  int cnt = 0;
  for (int t0 = 0; t0 < NTOK; t0 += 32) {
    const int t = t0 + lane;
    const int match = (expid[t] == e);
    const unsigned m = (unsigned)__ballot(match);               // wave32 mask
    const int pos = cnt + __popc(m & ((1u << lane) - 1u));
    if (match) {
      if (pos < CAP) {
        tok2slot[t] = pos;
        slot2tok[e * CAP + pos] = t;
      } else {
        tok2slot[t] = -1;                                       // dropped
      }
    }
    cnt += __popc(m);
  }
  if (lane == 0) {
    counts[e] = cnt;                                            // pre-drop count
    float me = me_sum[e] / (float)NTOK;
    float ce = (float)cnt / (float)NTOK;
    partial[e] = me * ce;
  }
  __syncthreads();
  if (tid == 0 && write_tail) {
    float l = 0.f;
#pragma unroll
    for (int c = 0; c < NEXP; ++c) l += partial[c];
    out_tail[0] = l * (float)NEXP;                              // l_aux
#pragma unroll
    for (int c = 0; c < NEXP; ++c) out_tail[1 + c] = (float)counts[c];
  }
}

// ------------- gather tokens into bf16 expert buffers -----------------------
__global__ __launch_bounds__(256)
void dispatch_kernel(const float* __restrict__ x, const int* __restrict__ slot2tok,
                     unsigned short* __restrict__ dispA) {
  const int slot = blockIdx.x;
  const int t = slot2tok[slot];
  if (t < 0) return;                                            // row pre-zeroed
  const float* src = x + (long)t * H_DIM;
  unsigned short* dst = dispA + (long)slot * H_DIM;
  const int i = threadIdx.x * 4;
  const float4 v = *(const float4*)(src + i);
  uint2 pk;
  pk.x = (unsigned)f2bf(v.x) | ((unsigned)f2bf(v.y) << 16);
  pk.y = (unsigned)f2bf(v.z) | ((unsigned)f2bf(v.w) << 16);
  *(uint2*)(dst + i) = pk;
}

// ---- bf16 WMMA GEMM: 128x128 block, BK=64, double-buffered LDS pipeline ----
// A: [M,K] row-major (f32 or bf16); B: [K,N] row-major f32 (converted to bf16
// while transposed into LDS as [N][K]); O: [M,N] f32 or bf16 (+ fused GELU).
// 8 waves of 64x32; 16 v_wmma per k-iteration behind a single barrier.
template<bool A_F32, bool DO_GELU, bool OUT_BF16>
__global__ __launch_bounds__(256)
void gemm_wmma(const void* __restrict__ Aall, const float* __restrict__ Ball,
               void* __restrict__ Oall, int M, int N, int K,
               long aStride, long bStride, long oStride) {
  __shared__ unsigned short ldsA[2][128 * 64];   // [buf][m][k]
  __shared__ unsigned short ldsB[2][128 * 64];   // [buf][n][k] (transposed)
  const int tid  = threadIdx.x;
  const int lane = tid & 31;
  const int wave = tid >> 5;
  const int wm = wave >> 2;                   // 0..1  (64-row slab)
  const int wn = wave & 3;                    // 0..3  (32-col slab)
  const int m0 = blockIdx.y * 128;
  const int n0 = blockIdx.x * 128;
  const int b  = blockIdx.z;

  const float* Af          = (const float*)Aall          + (A_F32 ? b * aStride : 0);
  const unsigned short* Ab = (const unsigned short*)Aall + (A_F32 ? 0 : b * aStride);
  const float* B           = Ball + b * bStride;
  float* Of          = (float*)Oall          + (OUT_BF16 ? 0 : b * oStride);
  unsigned short* Ob = (unsigned short*)Oall + (OUT_BF16 ? b * oStride : 0);

  f32x8 acc[4][2];
#pragma unroll
  for (int mi = 0; mi < 4; ++mi)
#pragma unroll
    for (int ni = 0; ni < 2; ++ni)
#pragma unroll
      for (int r = 0; r < 8; ++r) acc[mi][ni][r] = 0.f;

  const int arow = tid >> 1;                  // 0..127 (A row / B column)
  const int acb  = (tid & 1) * 32;            // 0 | 32 within BK=64

  // staging registers (one path dead per instantiation)
  float rAf[32];
  uint4 rAb[4];
  float rB[32];

  auto loadRegs = [&](int k0) {
    if (A_F32) {
      const float* s = Af + (long)(m0 + arow) * K + k0 + acb;
#pragma unroll
      for (int j = 0; j < 8; ++j)
        *(float4*)&rAf[j * 4] = *(const float4*)(s + j * 4);
    } else {
      const uint4* s = (const uint4*)(Ab + (long)(m0 + arow) * K + k0 + acb);
#pragma unroll
      for (int j = 0; j < 4; ++j) rAb[j] = s[j];
    }
    const float* sb = B + (long)(k0 + acb) * N + n0 + arow;     // column gather
#pragma unroll
    for (int j = 0; j < 32; ++j) rB[j] = sb[(long)j * N];
  };

  auto storeLds = [&](int p) {
    union { unsigned short s[32]; uint4 q[4]; } pk;
    if (A_F32) {
#pragma unroll
      for (int j = 0; j < 32; ++j) pk.s[j] = f2bf(rAf[j]);
    } else {
#pragma unroll
      for (int j = 0; j < 4; ++j) pk.q[j] = rAb[j];
    }
    uint4* da = (uint4*)&ldsA[p][arow * 64 + acb];
#pragma unroll
    for (int j = 0; j < 4; ++j) da[j] = pk.q[j];
    union { unsigned short s[32]; uint4 q[4]; } pb;
#pragma unroll
    for (int j = 0; j < 32; ++j) pb.s[j] = f2bf(rB[j]);
    uint4* db = (uint4*)&ldsB[p][arow * 64 + acb];
#pragma unroll
    for (int j = 0; j < 4; ++j) db[j] = pb.q[j];
  };

  // fragment layout per ISA 7.12.2 (16-bit 16x32 operands; B is N-major in
  // LDS so its per-lane layout matches A's): lanes 0-15 K{0..7,16..23},
  // lanes 16-31 K{8..15,24..31}, as two 16B ds_load_b128 each.
  const int fr = lane & 15;
  const int kb = (lane >> 4) * 8;

  auto compute = [&](int p) {
    union Frag { bf16x16 v; uint4 q[2]; };
#pragma unroll
    for (int ks = 0; ks < 64; ks += 32) {
      Frag a[4], bb[2];
#pragma unroll
      for (int mi = 0; mi < 4; ++mi) {
        const int row = (wm * 64 + mi * 16 + fr) * 64 + ks;
        a[mi].q[0] = *(const uint4*)&ldsA[p][row + kb];
        a[mi].q[1] = *(const uint4*)&ldsA[p][row + kb + 16];
      }
#pragma unroll
      for (int ni = 0; ni < 2; ++ni) {
        const int row = (wn * 32 + ni * 16 + fr) * 64 + ks;
        bb[ni].q[0] = *(const uint4*)&ldsB[p][row + kb];
        bb[ni].q[1] = *(const uint4*)&ldsB[p][row + kb + 16];
      }
#pragma unroll
      for (int mi = 0; mi < 4; ++mi)
#pragma unroll
        for (int ni = 0; ni < 2; ++ni)
          acc[mi][ni] = __builtin_amdgcn_wmma_f32_16x16x32_bf16(
              false, a[mi].v, false, bb[ni].v, (short)0, acc[mi][ni], false, false);
    }
  };

  // software pipeline: global loads for tile i+1 overlap WMMA on tile i
  loadRegs(0);
  storeLds(0);
  __syncthreads();
  int p = 0;
  for (int k0 = 0; k0 < K; k0 += 64) {
    const bool more = (k0 + 64) < K;
    if (more) loadRegs(k0 + 64);
    compute(p);
    if (more) storeLds(p ^ 1);
    __syncthreads();
    p ^= 1;
  }

  // epilogue: C/D layout — lanes 0-15 hold M=r, lanes 16-31 hold M=r+8; N=lane&15
  const int cn = lane & 15;
  const int rOff = (lane >> 4) * 8;
#pragma unroll
  for (int mi = 0; mi < 4; ++mi)
#pragma unroll
    for (int ni = 0; ni < 2; ++ni)
#pragma unroll
      for (int r = 0; r < 8; ++r) {
        const long gm = m0 + wm * 64 + mi * 16 + rOff + r;
        const long gn = n0 + wn * 32 + ni * 16 + cn;
        float v = acc[mi][ni][r];
        if (DO_GELU) v = gelu_tanh(v);
        if (OUT_BF16) Ob[gm * N + gn] = f2bf(v);
        else          Of[gm * N + gn] = v;
      }
}

// ------------- combine: out = c0*gate*expert_out + c1*res -------------------
__global__ __launch_bounds__(256)
void combine_kernel(const float* __restrict__ expert_out,
                    const int* __restrict__ expid, const int* __restrict__ tok2slot,
                    const float* __restrict__ gateprob, const float* __restrict__ coef0,
                    const float* __restrict__ coef1, float* __restrict__ out) {
  const long t = blockIdx.x;
  const int i = threadIdx.x * 4;
  const int e = expid[t];
  const int slot = tok2slot[t];
  const float g  = gateprob[t];
  const float c0 = coef0[t], c1 = coef1[t];
  float* op = out + t * H_DIM + i;
  const float4 r = *(const float4*)op;                 // residual written here
  float4 m = make_float4(0.f, 0.f, 0.f, 0.f);
  if (slot >= 0)
    m = *(const float4*)(expert_out + ((long)e * CAP + slot) * H_DIM + i);
  float4 o;
  o.x = c0 * g * m.x + c1 * r.x;
  o.y = c0 * g * m.y + c1 * r.y;
  o.z = c0 * g * m.z + c1 * r.z;
  o.w = c0 * g * m.w + c1 * r.w;
  *(float4*)op = o;
}

// ---------------------------------------------------------------------------
extern "C" void kernel_launch(void* const* d_in, const int* in_sizes, int n_in,
                              void* d_out, int out_size, void* d_ws, size_t ws_size,
                              hipStream_t stream) {
  const float* x      = (const float*)d_in[0];
  const float* wg     = (const float*)d_in[1];
  const float* w1     = (const float*)d_in[2];
  const float* w2     = (const float*)d_in[3];
  const float* res_w1 = (const float*)d_in[4];
  const float* res_w2 = (const float*)d_in[5];
  const float* coef_w = (const float*)d_in[6];
  const float* coef_b = (const float*)d_in[7];
  float* out = (float*)d_out;

  char* ws = (char*)d_ws;
  size_t off = 0;
  auto alloc = [&](size_t bytes) -> void* {
    void* p = ws + off;
    off = (off + bytes + 255) & ~(size_t)255;
    return p;
  };
  float* me_sum   = (float*)alloc(NEXP * sizeof(float));
  int*   expid    = (int*)alloc(NTOK * sizeof(int));
  float* gateprob = (float*)alloc(NTOK * sizeof(float));
  float* coef0    = (float*)alloc(NTOK * sizeof(float));
  float* coef1    = (float*)alloc(NTOK * sizeof(float));
  int*   tok2slot = (int*)alloc(NTOK * sizeof(int));
  int*   slot2tok = (int*)alloc((size_t)NEXP * CAP * sizeof(int));
  unsigned short* dispA = (unsigned short*)alloc((size_t)NEXP * CAP * H_DIM * 2);
  unsigned short* h_act = (unsigned short*)alloc((size_t)NEXP * CAP * FF_DIM * 2);
  float* expert_out     = (float*)alloc((size_t)NEXP * CAP * H_DIM * sizeof(float));
  unsigned short* res_h = (unsigned short*)alloc((size_t)NTOK * FF_DIM * 2);

  hipMemsetAsync(me_sum, 0, NEXP * sizeof(float), stream);
  hipMemsetAsync(slot2tok, 0xFF, (size_t)NEXP * CAP * sizeof(int), stream);      // -1
  hipMemsetAsync(dispA, 0, (size_t)NEXP * CAP * H_DIM * 2, stream);

  gate_kernel<<<NTOK / 8, 256, 0, stream>>>(x, wg, coef_w, coef_b,
                                            expid, gateprob, coef0, coef1, me_sum);
  const int write_tail = (out_size >= NTOK * H_DIM + 1 + NEXP) ? 1 : 0;
  assign_kernel<<<1, 256, 0, stream>>>(expid, me_sum, tok2slot, slot2tok,
                                       out + (size_t)NTOK * H_DIM, write_tail);
  dispatch_kernel<<<NEXP * CAP, 256, 0, stream>>>(x, slot2tok, dispA);

  // expert FC1: [C,H] x [H,FF] -> gelu -> bf16 h, batched over experts
  gemm_wmma<false, true, true><<<dim3(FF_DIM / 128, CAP / 128, NEXP), 256, 0, stream>>>(
      dispA, w1, h_act, CAP, FF_DIM, H_DIM,
      (long)CAP * H_DIM, (long)H_DIM * FF_DIM, (long)CAP * FF_DIM);
  // expert FC2: [C,FF] x [FF,H] -> f32 expert_out
  gemm_wmma<false, false, false><<<dim3(H_DIM / 128, CAP / 128, NEXP), 256, 0, stream>>>(
      h_act, w2, expert_out, CAP, H_DIM, FF_DIM,
      (long)CAP * FF_DIM, (long)FF_DIM * H_DIM, (long)CAP * H_DIM);
  // residual FC1: [n,H] x [H,FF] -> gelu -> bf16
  gemm_wmma<true, true, true><<<dim3(FF_DIM / 128, NTOK / 128, 1), 256, 0, stream>>>(
      x, res_w1, res_h, NTOK, FF_DIM, H_DIM, 0, 0, 0);
  // residual FC2: [n,FF] x [FF,H] -> f32, written directly into d_out
  gemm_wmma<false, false, false><<<dim3(H_DIM / 128, NTOK / 128, 1), 256, 0, stream>>>(
      res_h, res_w2, out, NTOK, H_DIM, FF_DIM, 0, 0, 0);

  combine_kernel<<<NTOK, 256, 0, stream>>>(expert_out, expid, tok2slot,
                                           gateprob, coef0, coef1, out);
}
